// HDClassifier_40956808134826
// MI455X (gfx1250) — matrix operand order repaired and checked by reading, not compiled, not run on hardware
//
#include <hip/hip_runtime.h>
#include <hip/hip_bf16.h>
#include <math.h>

// ---------------------------------------------------------------------------
// HD classifier for MI455X (gfx1250, wave32, WMMA).
//   scores = (enc @ cls^T) / (|enc_i| * |cls_c|)          [8192 x 100] f32
//   preds  = argmax(scores); wrong = preds != targets
//   new_hvs = cls + segsum(enc[wrong], targets) - segsum(enc[wrong], preds)
//
// GEMM pass: v_wmma_f32_16x16x32_f16 on f32->f16 staged LDS tiles, k-split
// 4-way across blocks (4096 waves) with f32 atomic combine of partial raw
// scores / row norms (L2-resident). Hot loop has NO bounds checks: only full
// 32-wide k-steps; the 16-wide remainder is a peeled tail step run by the
// last k-split block only. Padded B columns are pointer-clamped (their
// accumulator lanes are discarded), not zero-filled.
// ---------------------------------------------------------------------------

typedef _Float16 v8h  __attribute__((ext_vector_type(8)));
typedef _Float16 v16h __attribute__((ext_vector_type(16)));
typedef float    v8f  __attribute__((ext_vector_type(8)));

static constexpr int kN  = 8192;    // rows of encoded
static constexpr int kD  = 10000;   // hypervector dim
static constexpr int kC  = 100;     // classes
static constexpr int kCP = 128;     // classes padded to 8 WMMA col tiles
static constexpr int kBK = 32;      // k-step (WMMA f16 K)
static constexpr int kBR = 64;      // rows per workgroup
static constexpr int kKS = 4;       // k-split factor (grid.y)
static constexpr float kAlpha = 1.0f;

// ---------------------------------------------------------------------------
// Kernel 0: zero raw-score accumulator (in d_out) and row-sumsq accumulator
// (re-done every launch -> deterministic across graph replays)
// ---------------------------------------------------------------------------
__global__ __launch_bounds__(256) void hd_init(float* __restrict__ scores,
                                               float* __restrict__ rowss) {
  const int i = blockIdx.x * 256 + threadIdx.x;
  if (i < kN * kC) scores[i] = 0.f;
  if (i < kN) rowss[i] = 0.f;
}

// ---------------------------------------------------------------------------
// Kernel 1: class-HV L2 norms (tiny: 100 x 10000 reduction)
// ---------------------------------------------------------------------------
__global__ __launch_bounds__(256) void hd_cls_norms(const float* __restrict__ cls,
                                                    float* __restrict__ ncls) {
  __shared__ float red[256];
  const int c = blockIdx.x;
  float s = 0.f;
  for (int k = threadIdx.x; k < kD; k += 256) {
    float v = cls[(size_t)c * kD + k];
    s += v * v;
  }
  red[threadIdx.x] = s;
  __syncthreads();
  for (int off = 128; off > 0; off >>= 1) {
    if (threadIdx.x < off) red[threadIdx.x] += red[threadIdx.x + off];
    __syncthreads();
  }
  if (threadIdx.x == 0) ncls[c] = sqrtf(red[0]);
}

// ---------------------------------------------------------------------------
// Device helpers for the GEMM kernel
// ---------------------------------------------------------------------------
__device__ __forceinline__ void stage8_f16(_Float16* dst, float4 a, float4 b) {
  dst[0] = (_Float16)a.x;  dst[1] = (_Float16)a.y;
  dst[2] = (_Float16)a.z;  dst[3] = (_Float16)a.w;
  dst[4] = (_Float16)b.x;  dst[5] = (_Float16)b.y;
  dst[6] = (_Float16)b.z;  dst[7] = (_Float16)b.w;
}

// ---------------------------------------------------------------------------
// Kernel 2: WMMA partial GEMM over a k-slice + partial row sumsq.
// Grid: (kN/kBR) x kKS. 256 threads = 8 waves. Wave w: row tile (w&3) of 16
// rows, col group (w>>2) covering 4 x 16 padded class columns.
// Fragment layouts per CDNA5 ISA 7.12.2:
//   A (16x32 f16): lane m in [0,16): v0..3 = K0..7, v4..7 = K16..23 of row m;
//                  lane 16+m: K8..15 / K24..31 of row m.
//   B (32x16 f16): lane n in [0,16): K0..7 / K16..23 of col n; lane 16+n: K8..15 / K24..31.
// ---------------------------------------------------------------------------
__global__ __launch_bounds__(256) void hd_scores_wmma(const float* __restrict__ enc,
                                                      const float* __restrict__ cls,
                                                      float* __restrict__ scores,
                                                      float* __restrict__ rowss) {
  __shared__ __align__(16) _Float16 sA[kBR][kBK];   // 4 KB
  __shared__ __align__(16) _Float16 sB[kCP][kBK];   // 8 KB
  __shared__ float sRow[kBR][4];                    // per-row partial sumsq

  const int tid     = threadIdx.x;
  const int lane    = tid & 31;
  const int wave    = tid >> 5;
  const int rowTile = wave & 3;   // which 16-row tile of the 64-row block
  const int colGrp  = wave >> 2;  // 0: cols 0..63, 1: cols 64..127
  const int rowBase = blockIdx.x * kBR;

  const int nkFull = kD / kBK;                          // 312 full steps
  const int ksBeg  = (nkFull * (int)blockIdx.y) / kKS;  // 78-step slices
  const int ksEnd  = (nkFull * ((int)blockIdx.y + 1)) / kKS;
  const bool lastZ = ((int)blockIdx.y == kKS - 1);

  // A-staging mapping: thread -> (row = tid>>2, k-octet = (tid&3)*8)
  const int aRow = tid >> 2;
  const int aK   = (tid & 3) * 8;
  const float* aP = enc + (size_t)(rowBase + aRow) * kD + aK;

  // B-staging mapping: two reps of 64 columns, k-octet = (tid&3)*8.
  // Out-of-range padded columns (100..127) clamp to class kC-1: their data is
  // irrelevant (accumulator lanes for col >= kC are never stored) but loads
  // stay unconditional and in-bounds.
  const int bK    = (tid & 3) * 8;
  const int bCol0 = tid >> 2;          // 0..63
  const int bCol1 = bCol0 + 64;        // 64..127
  const int bCol1c = (bCol1 < kC) ? bCol1 : (kC - 1);
  const float* bP0 = cls + (size_t)bCol0 * kD + bK;
  const float* bP1 = cls + (size_t)bCol1c * kD + bK;

  v8f acc[4];
#pragma unroll
  for (int t = 0; t < 4; ++t)
    acc[t] = (v8f){0.f, 0.f, 0.f, 0.f, 0.f, 0.f, 0.f, 0.f};

  float sumsq = 0.f;

  // =================== hot loop: full 32-wide steps, no bounds checks ======
  for (int ks = ksBeg; ks < ksEnd; ++ks) {
    const int k0 = ks * kBK;

    // stage A tile (64 x 32) -> f16, accumulate row sumsq
    float4 x0 = *(const float4*)(aP + k0);
    float4 x1 = *(const float4*)(aP + k0 + 4);
    __builtin_prefetch(aP + k0 + kBK, 0, 3);   // global_prefetch_b8 (speculative-safe)
    sumsq += x0.x * x0.x + x0.y * x0.y + x0.z * x0.z + x0.w * x0.w
           + x1.x * x1.x + x1.y * x1.y + x1.z * x1.z + x1.w * x1.w;
    stage8_f16(&sA[aRow][aK], x0, x1);

    // stage B tile (128 x 32) -> f16 (cols >= kC carry clamped/ignored data)
    float4 y0 = *(const float4*)(bP0 + k0);
    float4 y1 = *(const float4*)(bP0 + k0 + 4);
    float4 z0 = *(const float4*)(bP1 + k0);
    float4 z1 = *(const float4*)(bP1 + k0 + 4);
    stage8_f16(&sB[bCol0][bK], y0, y1);
    stage8_f16(&sB[bCol1][bK], z0, z1);
    __syncthreads();

    // build fragments from LDS, 4 WMMAs per wave
    const int frow = rowTile * 16 + (lane & 15);
    const int akb  = (lane < 16) ? 0 : 8;
    v8h alo = *(const v8h*)&sA[frow][akb];
    v8h ahi = *(const v8h*)&sA[frow][akb + 16];
    v16h afrag = __builtin_shufflevector(alo, ahi, 0, 1, 2, 3, 4, 5, 6, 7,
                                         8, 9, 10, 11, 12, 13, 14, 15);
    const int bkb = (lane < 16) ? 0 : 8;
#pragma unroll
    for (int t = 0; t < 4; ++t) {
      const int bc = colGrp * 64 + t * 16 + (lane & 15);
      v8h blo = *(const v8h*)&sB[bc][bkb];
      v8h bhi = *(const v8h*)&sB[bc][bkb + 16];
      v16h bfrag = __builtin_shufflevector(blo, bhi, 0, 1, 2, 3, 4, 5, 6, 7,
                                           8, 9, 10, 11, 12, 13, 14, 15);
      acc[t] = __builtin_amdgcn_wmma_f32_16x16x32_f16(
          false, afrag, false, bfrag, (short)0, acc[t], false, false);
    }
    __syncthreads();
  }

  // =================== peeled tail: 16 valid k, last k-split block only ====
  if (lastZ) {
    const int k0 = nkFull * kBK;                 // 9984
    const float4 zf4 = make_float4(0.f, 0.f, 0.f, 0.f);
    const bool kv = (aK < 16);                   // octets 0,8 valid; 16,24 not
    float4 x0 = kv ? *(const float4*)(aP + k0)     : zf4;
    float4 x1 = kv ? *(const float4*)(aP + k0 + 4) : zf4;
    sumsq += x0.x * x0.x + x0.y * x0.y + x0.z * x0.z + x0.w * x0.w
           + x1.x * x1.x + x1.y * x1.y + x1.z * x1.z + x1.w * x1.w;
    stage8_f16(&sA[aRow][aK], x0, x1);

    const bool kvb = (bK < 16);
    float4 y0 = kvb ? *(const float4*)(bP0 + k0)     : zf4;
    float4 y1 = kvb ? *(const float4*)(bP0 + k0 + 4) : zf4;
    float4 z0 = kvb ? *(const float4*)(bP1 + k0)     : zf4;
    float4 z1 = kvb ? *(const float4*)(bP1 + k0 + 4) : zf4;
    stage8_f16(&sB[bCol0][bK], y0, y1);
    stage8_f16(&sB[bCol1][bK], z0, z1);
    __syncthreads();

    const int frow = rowTile * 16 + (lane & 15);
    const int akb  = (lane < 16) ? 0 : 8;
    v8h alo = *(const v8h*)&sA[frow][akb];
    v8h ahi = *(const v8h*)&sA[frow][akb + 16];
    v16h afrag = __builtin_shufflevector(alo, ahi, 0, 1, 2, 3, 4, 5, 6, 7,
                                         8, 9, 10, 11, 12, 13, 14, 15);
    const int bkb = (lane < 16) ? 0 : 8;
#pragma unroll
    for (int t = 0; t < 4; ++t) {
      const int bc = colGrp * 64 + t * 16 + (lane & 15);
      v8h blo = *(const v8h*)&sB[bc][bkb];
      v8h bhi = *(const v8h*)&sB[bc][bkb + 16];
      v16h bfrag = __builtin_shufflevector(blo, bhi, 0, 1, 2, 3, 4, 5, 6, 7,
                                           8, 9, 10, 11, 12, 13, 14, 15);
      acc[t] = __builtin_amdgcn_wmma_f32_16x16x32_f16(
          false, afrag, false, bfrag, (short)0, acc[t], false, false);
    }
    __syncthreads();
  }

  // ---- combine partial row sumsq ------------------------------------------
  sRow[aRow][tid & 3] = sumsq;
  __syncthreads();
  if (tid < kBR) {
    atomicAdd(&rowss[rowBase + tid],
              sRow[tid][0] + sRow[tid][1] + sRow[tid][2] + sRow[tid][3]);
  }

  // ---- combine partial raw scores -----------------------------------------
#pragma unroll
  for (int t = 0; t < 4; ++t) {
    const int col = colGrp * 64 + t * 16 + (lane & 15);
    if (col < kC) {
#pragma unroll
      for (int r = 0; r < 8; ++r) {
        const int rloc = rowTile * 16 + r + ((lane < 16) ? 0 : 8);
        atomicAdd(&scores[(size_t)(rowBase + rloc) * kC + col], acc[t][r]);
      }
    }
  }
}

// ---------------------------------------------------------------------------
// Kernel 3: fused normalize (in place) + row-wise argmax (wave per row)
// -> pred if wrong else -1
// ---------------------------------------------------------------------------
__global__ __launch_bounds__(256) void hd_norm_argmax(float* __restrict__ scores,
                                                      const float* __restrict__ rowss,
                                                      const float* __restrict__ ncls,
                                                      const int* __restrict__ targets,
                                                      int* __restrict__ predw) {
  const int row  = blockIdx.x * 8 + (threadIdx.x >> 5);
  const int lane = threadIdx.x & 31;
  const float ine = 1.0f / sqrtf(rowss[row]);
  float best = -3.402823e38f;
  int   bi   = 0x7fffffff;
  for (int c = lane; c < kC; c += 32) {
    float v = scores[(size_t)row * kC + c] * (ine / ncls[c]);
    scores[(size_t)row * kC + c] = v;       // in-place normalized score
    if (v > best || (v == best && c < bi)) { best = v; bi = c; }
  }
#pragma unroll
  for (int off = 16; off > 0; off >>= 1) {
    const float ov = __shfl_xor(best, off, 32);
    const int   oi = __shfl_xor(bi, off, 32);
    if (ov > best || (ov == best && oi < bi)) { best = ov; bi = oi; }
  }
  if (lane == 0) {
    const int t = targets[row];
    predw[row] = (bi != t) ? bi : -1;
  }
}

// ---------------------------------------------------------------------------
// Kernel 4: perceptron update. Block = (d-chunk of 128, row-group of 1024).
// Each thread owns one d-column: race-free LDS accumulate over classes,
// then atomicAdd into new_class_hvs (pre-initialized with class_hvs).
// ---------------------------------------------------------------------------
static constexpr int kDC = 128;      // d-columns per block
static constexpr int kRG = kN / 8;   // rows per row-group (1024)

__global__ __launch_bounds__(128) void hd_update(const float* __restrict__ enc,
                                                 const int* __restrict__ targets,
                                                 const int* __restrict__ predw,
                                                 float* __restrict__ newhvs) {
  __shared__ float acc[kC * kDC];    // 51.2 KB
  const int tid   = threadIdx.x;
  const int d     = blockIdx.x * kDC + tid;
  const bool dok  = d < kD;

  for (int i = tid; i < kC * kDC; i += kDC) acc[i] = 0.f;
  __syncthreads();

  const int r0 = blockIdx.y * kRG;
  for (int i = r0; i < r0 + kRG; ++i) {
    const int p = predw[i];          // uniform across block
    if (p < 0) continue;             // correctly classified: no update
    const int t = targets[i];
    const float v = dok ? enc[(size_t)i * kD + d] : 0.f;
    acc[t * kDC + tid] += v * (kAlpha * kAlpha);  // add_t (ALPHA applied twice in ref)
    acc[p * kDC + tid] -= v * kAlpha;             // sub_p * ALPHA
  }
  __syncthreads();

  if (dok) {
    for (int c = 0; c < kC; ++c)
      atomicAdd(&newhvs[(size_t)c * kD + d], acc[c * kDC + tid]);
  }
}

// ---------------------------------------------------------------------------
// Host-side launcher
// ---------------------------------------------------------------------------
extern "C" void kernel_launch(void* const* d_in, const int* in_sizes, int n_in,
                              void* d_out, int out_size, void* d_ws, size_t ws_size,
                              hipStream_t stream) {
  const float* enc     = (const float*)d_in[0];   // [8192, 10000]
  const float* cls     = (const float*)d_in[1];   // [100, 10000]
  const int*   targets = (const int*)d_in[2];     // [8192]

  float* scores = (float*)d_out;                       // [8192, 100]
  float* newhvs = scores + (size_t)kN * kC;            // [100, 10000]

  float* ncls  = (float*)d_ws;                         // 100 floats
  int*   predw = (int*)((char*)d_ws + 512);            // 8192 ints (32 KB)
  float* rowss = (float*)((char*)d_ws + 512 + kN * sizeof(int));  // 8192 floats

  // 0) zero accumulators (every launch -> deterministic replays)
  hd_init<<<(kN * kC + 255) / 256, 256, 0, stream>>>(scores, rowss);

  // 1) class norms
  hd_cls_norms<<<kC, 256, 0, stream>>>(cls, ncls);

  // 2) init new_class_hvs = class_hvs
  hipMemcpyAsync(newhvs, cls, sizeof(float) * (size_t)kC * kD,
                 hipMemcpyDeviceToDevice, stream);

  // 3) WMMA partial GEMMs, k-split 4-way for occupancy (512 WGs, 4096 waves)
  hd_scores_wmma<<<dim3(kN / kBR, kKS), 256, 0, stream>>>(enc, cls, scores, rowss);

  // 4) fused normalize + argmax -> pred-if-wrong
  hd_norm_argmax<<<kN / 8, 256, 0, stream>>>(scores, rowss, ncls, targets, predw);

  // 5) scatter update into new_class_hvs
  hd_update<<<dim3((kD + kDC - 1) / kDC, kN / kRG), kDC, 0, stream>>>(
      enc, targets, predw, newhvs);
}
